// ConvPointnetLite_52493090292351
// MI455X (gfx1250) — compile-verified, matrix operand order checked by reading, not run on hardware
//
#include <hip/hip_runtime.h>
#include <hip/hip_bf16.h>

typedef _Float16 v8h  __attribute__((ext_vector_type(8)));
typedef _Float16 v16h __attribute__((ext_vector_type(16)));
typedef float    v8f  __attribute__((ext_vector_type(8)));

#define BB   4
#define NP   30000
#define NQ   30000
#define TOT  (BB * NP)          // 120000 points
#define R2   4096               // 64*64 bins
#define FCH  256
#define PW   66                 // padded grid width (1-pixel zero halo)
#define PP   (PW * PW)          // 4356 padded pixels

// ---------------------------------------------------------------------------
// Pack conv weights (O,I,3,3) f32 -> (tap, o, i) f16 so A-fragments are
// contiguous-in-K per lane.
// ---------------------------------------------------------------------------
__global__ void k_pack_w(const float* __restrict__ W, _Float16* __restrict__ Wp) {
  long i = (long)blockIdx.x * 256 + threadIdx.x;   // 9*256*256 = 589824
  if (i >= 9L * 256 * 256) return;
  int ii  = (int)(i & 255);
  long r  = i >> 8;
  int o   = (int)(r & 255);
  int tap = (int)(r >> 8);
  int ky = tap / 3, kx = tap % 3;
  Wp[i] = (_Float16)W[(((long)o * 256 + ii) * 3 + ky) * 3 + kx];
}

// ---------------------------------------------------------------------------
// Stage 0: net0 = fc_pos(p); resblock0 -> net[64]; plane bin indices.
// Block = 256 threads, 64 points per block, LDS-tiled GEMV phases.
// ---------------------------------------------------------------------------
__global__ void k_stage0(const float* __restrict__ p,
                         const float* __restrict__ fcW, const float* __restrict__ fcb,
                         const float* __restrict__ f0W, const float* __restrict__ f0b,
                         const float* __restrict__ f1W, const float* __restrict__ f1b,
                         const float* __restrict__ scW,
                         float* __restrict__ netOut, int* __restrict__ idxOut) {
  __shared__ float s_net0[64][128];
  __shared__ float s_h[64][64];
  __shared__ float s_p[64][3];
  int t = threadIdx.x;
  long base = (long)blockIdx.x * 64;

  for (int i = t; i < 64 * 3; i += 256) {
    int pt = i / 3, c = i % 3;
    long g = base + pt;
    s_p[pt][c] = (g < TOT) ? p[g * 3 + c] : 0.f;
  }
  __syncthreads();

  { // fc_pos: 3 -> 128
    int j = t & 127, ptb = t >> 7;
    float w0 = fcW[j * 3 + 0], w1 = fcW[j * 3 + 1], w2 = fcW[j * 3 + 2], b0 = fcb[j];
    for (int r = 0; r < 32; ++r) {
      int pt = ptb * 32 + r;
      s_net0[pt][j] = b0 + w0 * s_p[pt][0] + w1 * s_p[pt][1] + w2 * s_p[pt][2];
    }
  }
  __syncthreads();

  { // h = relu(fc0 @ net0)
    int o = t & 63, ptb = t >> 6;
    const float* wr = f0W + o * 128;
    float bo = f0b[o];
    for (int r = 0; r < 16; ++r) {
      int pt = ptb * 16 + r;
      float acc = bo;
      for (int j = 0; j < 128; ++j) acc += wr[j] * s_net0[pt][j];
      s_h[pt][o] = fmaxf(acc, 0.f);
    }
  }
  __syncthreads();

  { // out = sc(net0) + relu(fc1 @ h)
    int o = t & 63, ptb = t >> 6;
    const float* w1r = f1W + o * 64;
    const float* wsr = scW + o * 128;
    float b1o = f1b[o];
    for (int r = 0; r < 16; ++r) {
      int pt = ptb * 16 + r;
      long g = base + pt;
      if (g >= TOT) continue;
      float dx = b1o;
      for (int j = 0; j < 64; ++j) dx += w1r[j] * s_h[pt][j];
      dx = fmaxf(dx, 0.f);
      float sc = 0.f;
      for (int j = 0; j < 128; ++j) sc += wsr[j] * s_net0[pt][j];
      netOut[g * 64 + o] = sc + dx;
    }
  }

  // plane bin indices: plane0=xz(0,2), plane1=xy(0,1), plane2=yz(1,2)
  if (t < 192) {
    int pt = t & 63, pl = t >> 6;
    long g = base + pt;
    if (g < TOT) {
      const int axA[3] = {0, 0, 1}, axB[3] = {2, 1, 2};
      const float inv = 1.0f / 1.100001f;       // 1 + PAD + 1e-6
      float na = fminf(fmaxf((s_p[pt][axA[pl]] + 0.5f) * inv, 0.f), 0.999999f);
      float nb = fminf(fmaxf((s_p[pt][axB[pl]] + 0.5f) * inv, 0.f), 0.999999f);
      int xa = (int)(na * 64.f); if (xa > 63) xa = 63;
      int xb = (int)(nb * 64.f); if (xb > 63) xb = 63;
      idxOut[(long)pl * TOT + g] = xa + 64 * xb;
    }
  }
}

// ---------------------------------------------------------------------------
// Scatter-add 64-ch features into xz bins.
// ---------------------------------------------------------------------------
__global__ void k_scatter64(const float* __restrict__ net, const int* __restrict__ idx0,
                            float* __restrict__ binsum, float* __restrict__ bincnt) {
  long g = (long)blockIdx.x * 256 + threadIdx.x;
  if (g >= TOT) return;
  int b = (int)(g / NP);
  long bin = (long)b * R2 + idx0[g];
  const float* src = net + g * 64;
  for (int c = 0; c < 64; ++c) atomicAdd(&binsum[bin * 64 + c], src[c]);
  atomicAdd(&bincnt[bin], 1.f);
}

// ---------------------------------------------------------------------------
// Gather pooled mean, concat with net, resblock -> new net[64].
// ---------------------------------------------------------------------------
__global__ void k_pool_res(const float* __restrict__ netIn,
                           const float* __restrict__ binsum, const float* __restrict__ bincnt,
                           const int* __restrict__ idx0,
                           const float* __restrict__ f0W, const float* __restrict__ f0b,
                           const float* __restrict__ f1W, const float* __restrict__ f1b,
                           const float* __restrict__ scW, float* __restrict__ netOut) {
  __shared__ float s_in[64][128];
  __shared__ float s_h[64][64];
  int t = threadIdx.x;
  long base = (long)blockIdx.x * 64;

  for (int i = t; i < 64 * 128; i += 256) {
    int pt = i >> 7, c = i & 127;
    long g = base + pt;
    float v = 0.f;
    if (g < TOT) {
      if (c < 64) v = netIn[g * 64 + c];
      else {
        int b = (int)(g / NP);
        long bin = (long)b * R2 + idx0[g];
        v = binsum[bin * 64 + (c - 64)] / fmaxf(bincnt[bin], 1.f);
      }
    }
    s_in[pt][c] = v;
  }
  __syncthreads();

  { int o = t & 63, ptb = t >> 6;
    const float* wr = f0W + o * 128;
    float bo = f0b[o];
    for (int r = 0; r < 16; ++r) {
      int pt = ptb * 16 + r;
      float acc = bo;
      for (int j = 0; j < 128; ++j) acc += wr[j] * s_in[pt][j];
      s_h[pt][o] = fmaxf(acc, 0.f);
    }
  }
  __syncthreads();

  { int o = t & 63, ptb = t >> 6;
    const float* w1r = f1W + o * 64;
    const float* wsr = scW + o * 128;
    float b1o = f1b[o];
    for (int r = 0; r < 16; ++r) {
      int pt = ptb * 16 + r;
      long g = base + pt;
      if (g >= TOT) continue;
      float dx = b1o;
      for (int j = 0; j < 64; ++j) dx += w1r[j] * s_h[pt][j];
      dx = fmaxf(dx, 0.f);
      float sc = 0.f;
      for (int j = 0; j < 128; ++j) sc += wsr[j] * s_in[pt][j];
      netOut[g * 64 + o] = sc + dx;
    }
  }
}

// ---------------------------------------------------------------------------
// c = fc_c(net) [256]; scatter-add into the 3 plane grids.
// Block = 256 threads (one per channel), 64 points per block.
// ---------------------------------------------------------------------------
__global__ void k_fcc_scatter(const float* __restrict__ net, const int* __restrict__ idx,
                              const float* __restrict__ fcW, const float* __restrict__ fcb,
                              float* __restrict__ planeSum, float* __restrict__ planeCnt) {
  __shared__ float s_net[64][64];
  __shared__ int   s_idx[3][64];
  __shared__ int   s_bat[64];
  int t = threadIdx.x;
  long base = (long)blockIdx.x * 64;

  for (int i = t; i < 64 * 64; i += 256) {
    int pt = i >> 6, c = i & 63;
    long g = base + pt;
    s_net[pt][c] = (g < TOT) ? net[g * 64 + c] : 0.f;
  }
  if (t < 192) {
    int pt = t & 63, pl = t >> 6;
    long g = base + pt;
    s_idx[pl][pt] = (g < TOT) ? idx[(long)pl * TOT + g] : 0;
  }
  if (t < 64) {
    long g = base + t;
    s_bat[t] = (g < TOT) ? (int)(g / NP) : 0;
  }
  __syncthreads();

  float wrow[64];
#pragma unroll
  for (int j = 0; j < 64; ++j) wrow[j] = fcW[t * 64 + j];
  float bias = fcb[t];

  for (int pt = 0; pt < 64; ++pt) {
    long g = base + pt;
    if (g >= TOT) break;
    float acc = bias;
#pragma unroll
    for (int j = 0; j < 64; ++j) acc += wrow[j] * s_net[pt][j];
    int b = s_bat[pt];
#pragma unroll
    for (int pl = 0; pl < 3; ++pl)
      atomicAdd(&planeSum[(((long)(pl * BB + b)) * R2 + s_idx[pl][pt]) * FCH + t], acc);
  }
  if (t < 3) {
    for (int pt = 0; pt < 64; ++pt) {
      long g = base + pt;
      if (g >= TOT) break;
      atomicAdd(&planeCnt[((long)(t * BB + s_bat[pt])) * R2 + s_idx[t][pt]], 1.f);
    }
  }
}

// ---------------------------------------------------------------------------
// planeSum/cnt -> f16 grid features in the ZERO-PADDED halo layout
// [pb][y+1][x+1][ch] with row stride PW=66. Halo stays 0 from the memset.
// ---------------------------------------------------------------------------
__global__ void k_mean2half(const float* __restrict__ planeSum, const float* __restrict__ planeCnt,
                            _Float16* __restrict__ feaH) {
  long bin = blockIdx.x;                 // pb*R2 + (y*64+x), 3*B*R2 bins
  int t = threadIdx.x;                   // 256 channels
  long pbb = bin >> 12;
  int pix = (int)(bin & (R2 - 1));
  int yy = pix >> 6, xx = pix & 63;
  float inv = 1.f / fmaxf(planeCnt[bin], 1.f);
  long dst = (pbb * PP + (long)(yy + 1) * PW + (xx + 1)) * FCH + t;
  feaH[dst] = (_Float16)(planeSum[bin * FCH + t] * inv);
}

// ---------------------------------------------------------------------------
// 3x3 conv (256->256) + bias + ReLU as a 9-tap WMMA GEMM on the padded grid.
// Block = 512 threads (16 waves): one 16-pixel tile, all 16 o-tiles.
// The 3x18x256 f16 input halo (27.6KB) is staged once in LDS; every wave
// reads B fragments with ds_load_b128, A fragments stream from L2.
// No border predication anywhere: halo zeros live in memory.
// grid = (256 pixel-tiles, 1, 12 plane*batch).
// ---------------------------------------------------------------------------
__global__ __launch_bounds__(512)
void k_conv3x3(const _Float16* __restrict__ inF, const _Float16* __restrict__ Wp,
               const float* __restrict__ bias,
               _Float16* __restrict__ outH, float* __restrict__ outF) {
  __shared__ __align__(16) _Float16 s_tile[3 * 18 * 256];   // 27648 B of 320KB WGP LDS

  int t = threadIdx.x;
  int wave = t >> 5, lane = t & 31;
  int n = lane & 15, hsel = lane >> 4;
  int pixTile = blockIdx.x;              // 0..255
  int y  = pixTile >> 2;                 // 0..63
  int x0 = (pixTile & 3) << 4;           // 0,16,32,48
  long pb = blockIdx.z;
  const _Float16* base = inF + pb * (long)PP * FCH;

  // Stage input halo: rows y-1..y+1 (padded rows y..y+2), cols x0-1..x0+16
  // (padded cols x0..x0+17); 8-half (16B) units.
  for (int u = t; u < 3 * 18 * 32; u += 512) {
    int pixl = u >> 5;                   // 0..53: r*18 + cLoc
    int k8 = (u & 31) << 3;
    int r = pixl / 18, cLoc = pixl % 18;
    long srcPix = (long)(y + r) * PW + (x0 + cLoc);
    *(v8h*)&s_tile[pixl * 256 + k8] = *(const v8h*)(base + srcPix * FCH + k8);
  }
  __syncthreads();

  int o0 = wave << 4;
  v8f acc = {};
  for (int tap = 0; tap < 9; ++tap) {
    int dy = tap / 3, dx = tap % 3;      // 0..2
    const _Float16* wrow = Wp + ((long)(tap * 256 + (o0 + n))) * 256 + 8 * hsel;
    const _Float16* brow = &s_tile[((dy * 18) + (n + dx)) * 256 + 8 * hsel];
#pragma unroll
    for (int k0 = 0; k0 < 256; k0 += 32) {
      v8h a0 = *(const v8h*)(wrow + k0);
      v8h a1 = *(const v8h*)(wrow + k0 + 16);
      v8h b0 = *(const v8h*)(brow + k0);
      v8h b1 = *(const v8h*)(brow + k0 + 16);
      v16h A  = __builtin_shufflevector(a0, a1, 0,1,2,3,4,5,6,7,8,9,10,11,12,13,14,15);
      v16h Bv = __builtin_shufflevector(b0, b1, 0,1,2,3,4,5,6,7,8,9,10,11,12,13,14,15);
      acc = __builtin_amdgcn_wmma_f32_16x16x32_f16(false, A, false, Bv,
                                                   (short)0, acc, false, false);
    }
  }

  int xcol = x0 + n;                     // 0..63
#pragma unroll
  for (int r = 0; r < 8; ++r) {
    int o = o0 + r + 8 * hsel;           // C/D layout: M = r + 8*(lane>>4)
    float v = fmaxf(acc[r] + bias[o], 0.f);
    if (outH) {                          // conv1: padded f16 layout
      long off = (pb * PP + (long)(y + 1) * PW + (xcol + 1)) * FCH + o;
      outH[off] = (_Float16)v;
    } else {                             // conv2: dense f32 layout for sampling
      long off = (pb * R2 + (long)y * 64 + xcol) * FCH + o;
      outF[off] = v;
    }
  }
}

// ---------------------------------------------------------------------------
// Bilinear grid sample (border pad, align_corners=False) + 3-plane sum.
// Block = 256 threads (one per channel), 4 queries per block.
// ---------------------------------------------------------------------------
__global__ void k_gridsample(const float* __restrict__ query, const float* __restrict__ fea,
                             float* __restrict__ out) {
  int t = threadIdx.x;
  for (int qi = 0; qi < 4; ++qi) {
    long q = (long)blockIdx.x * 4 + qi;
    if (q >= (long)BB * NQ) return;
    int b = (int)(q / NQ);
    float q0 = query[q * 3 + 0], q1 = query[q * 3 + 1], q2 = query[q * 3 + 2];
    float acc = 0.f;
#pragma unroll
    for (int pl = 0; pl < 3; ++pl) {
      float ga = (pl == 2) ? q1 : q0;                // plane coord a
      float gb = (pl == 1) ? q1 : q2;                // plane coord b
      float x = fminf(fmaxf(ga * 64.f - 0.5f, 0.f), 63.f);
      float y = fminf(fmaxf(gb * 64.f - 0.5f, 0.f), 63.f);
      float xf = floorf(x), yf = floorf(y);
      float wx = x - xf, wy = y - yf;
      int x0i = (int)xf, y0i = (int)yf;
      int x1i = min(x0i + 1, 63), y1i = min(y0i + 1, 63);
      const float* Fp = fea + ((long)(pl * BB + b)) * R2 * FCH;
      float f00 = Fp[((long)y0i * 64 + x0i) * FCH + t];
      float f01 = Fp[((long)y0i * 64 + x1i) * FCH + t];
      float f10 = Fp[((long)y1i * 64 + x0i) * FCH + t];
      float f11 = Fp[((long)y1i * 64 + x1i) * FCH + t];
      acc += (1.f - wx) * (1.f - wy) * f00 + wx * (1.f - wy) * f01
           + (1.f - wx) * wy * f10 + wx * wy * f11;
    }
    out[q * FCH + t] = acc;
  }
}

// ---------------------------------------------------------------------------
extern "C" void kernel_launch(void* const* d_in, const int* in_sizes, int n_in,
                              void* d_out, int out_size, void* d_ws, size_t ws_size,
                              hipStream_t stream) {
  (void)in_sizes; (void)n_in; (void)out_size; (void)ws_size;
  const float* p        = (const float*)d_in[0];
  const float* query    = (const float*)d_in[1];
  const float* fc_pos_W = (const float*)d_in[2];
  const float* fc_pos_b = (const float*)d_in[3];
  const float* b_f0W    = (const float*)d_in[4];   // [3,64,128]
  const float* b_f0b    = (const float*)d_in[5];   // [3,64]
  const float* b_f1W    = (const float*)d_in[6];   // [3,64,64]
  const float* b_f1b    = (const float*)d_in[7];   // [3,64]
  const float* b_scW    = (const float*)d_in[8];   // [3,64,128]
  const float* fc_c_W   = (const float*)d_in[9];
  const float* fc_c_b   = (const float*)d_in[10];
  const float* conv1_W  = (const float*)d_in[11];
  const float* conv1_b  = (const float*)d_in[12];
  const float* conv2_W  = (const float*)d_in[13];
  const float* conv2_b  = (const float*)d_in[14];
  float* out = (float*)d_out;

  // Workspace layout (256B aligned), ~175 MB total.
  char* ws = (char*)d_ws;
  size_t off = 0;
  auto take = [&](size_t bytes) { size_t o = off; off = (off + bytes + 255) & ~(size_t)255; return o; };
  float*    net0     = (float*)(ws + take((size_t)TOT * 64 * 4));
  float*    net1     = (float*)(ws + take((size_t)TOT * 64 * 4));
  int*      idx3     = (int*)  (ws + take((size_t)3 * TOT * 4));
  float*    binsum   = (float*)(ws + take((size_t)BB * R2 * 64 * 4));
  float*    bincnt   = (float*)(ws + take((size_t)BB * R2 * 4));
  float*    planeSum = (float*)(ws + take((size_t)3 * BB * R2 * FCH * 4)); // reused as conv2 f32 out
  float*    planeCnt = (float*)(ws + take((size_t)3 * BB * R2 * 4));
  _Float16* feaH     = (_Float16*)(ws + take((size_t)3 * BB * PP * FCH * 2)); // padded halo layout
  _Float16* conv1H   = (_Float16*)(ws + take((size_t)3 * BB * PP * FCH * 2)); // padded halo layout
  _Float16* Wp1      = (_Float16*)(ws + take((size_t)9 * 256 * 256 * 2));
  _Float16* Wp2      = (_Float16*)(ws + take((size_t)9 * 256 * 256 * 2));
  float*    conv2F   = planeSum;   // alias: planeSum is dead after k_mean2half

  // Zero accumulators + halo buffers (graph-capturable async memsets).
  hipMemsetAsync(planeSum, 0, (size_t)3 * BB * R2 * FCH * 4, stream);
  hipMemsetAsync(planeCnt, 0, (size_t)3 * BB * R2 * 4, stream);
  hipMemsetAsync(binsum,   0, (size_t)BB * R2 * 64 * 4, stream);
  hipMemsetAsync(bincnt,   0, (size_t)BB * R2 * 4, stream);
  hipMemsetAsync(feaH,     0, (size_t)3 * BB * PP * FCH * 2, stream);
  hipMemsetAsync(conv1H,   0, (size_t)3 * BB * PP * FCH * 2, stream);

  // Pack conv weights for WMMA fragment loads.
  k_pack_w<<<2304, 256, 0, stream>>>(conv1_W, Wp1);
  k_pack_w<<<2304, 256, 0, stream>>>(conv2_W, Wp2);

  // Point MLP stage 0 + bin indices.
  k_stage0<<<TOT / 64, 256, 0, stream>>>(p, fc_pos_W, fc_pos_b,
                                         b_f0W + 0 * 64 * 128, b_f0b + 0 * 64,
                                         b_f1W + 0 * 64 * 64,  b_f1b + 0 * 64,
                                         b_scW + 0 * 64 * 128, net0, idx3);

  // Pooling round 1 (resblock 1).
  k_scatter64<<<(TOT + 255) / 256, 256, 0, stream>>>(net0, idx3, binsum, bincnt);
  k_pool_res<<<TOT / 64, 256, 0, stream>>>(net0, binsum, bincnt, idx3,
                                           b_f0W + 1 * 64 * 128, b_f0b + 1 * 64,
                                           b_f1W + 1 * 64 * 64,  b_f1b + 1 * 64,
                                           b_scW + 1 * 64 * 128, net1);

  // Pooling round 2 (resblock 2).
  hipMemsetAsync(binsum, 0, (size_t)BB * R2 * 64 * 4, stream);
  hipMemsetAsync(bincnt, 0, (size_t)BB * R2 * 4, stream);
  k_scatter64<<<(TOT + 255) / 256, 256, 0, stream>>>(net1, idx3, binsum, bincnt);
  k_pool_res<<<TOT / 64, 256, 0, stream>>>(net1, binsum, bincnt, idx3,
                                           b_f0W + 2 * 64 * 128, b_f0b + 2 * 64,
                                           b_f1W + 2 * 64 * 64,  b_f1b + 2 * 64,
                                           b_scW + 2 * 64 * 128, net0);

  // fc_c + scatter into 3 plane grids, then mean -> padded f16 grids.
  k_fcc_scatter<<<TOT / 64, 256, 0, stream>>>(net0, idx3, fc_c_W, fc_c_b, planeSum, planeCnt);
  k_mean2half<<<3 * BB * R2, 256, 0, stream>>>(planeSum, planeCnt, feaH);

  // Two 3x3 convs on the WMMA path (LDS-staged B, halo-padded, branch-free).
  dim3 cgrid(256, 1, 3 * BB);
  k_conv3x3<<<cgrid, 512, 0, stream>>>(feaH,   Wp1, conv1_b, conv1H, nullptr);
  k_conv3x3<<<cgrid, 512, 0, stream>>>(conv1H, Wp2, conv2_b, nullptr, conv2F);

  // Bilinear sample + plane sum -> output [B,Nq,256].
  k_gridsample<<<(BB * NQ) / 4, 256, 0, stream>>>(query, conv2F, out);
}
